// CRF_19000935317717
// MI455X (gfx1250) — matrix operand order repaired and checked by reading, not compiled, not run on hardware
//
#include <hip/hip_runtime.h>

// CRF forward (log-partition), T=256, B=128, N=64.
// State kept in linear domain: alpha[b][:] = ln2 * (C_b + log2 w[b][:]),
// with w row-normalized (max = 1) and stored f16 -> directly usable as the
// WMMA A operand. Per step:  q = w x E  (2x v_wmma_f32_16x16x32_f16, K=64),
// v = q * 2^(unary*log2e),  w' = v / rowmax(v),  C_b += log2(rowmax(v)).
// Frozen rows (t >= len) skip the commit, preserving w and C_b exactly.

#define CRF_T 256
#define CRF_B 128
#define CRF_N 64
#define START_IDX 1
#define END_IDX 2
#define LOG2E 1.4426950408889634f
#define LN2   0.6931471805599453f

typedef __attribute__((ext_vector_type(16))) _Float16 v16h;
typedef __attribute__((ext_vector_type(8)))  float    v8f;

__global__ __launch_bounds__(128) void crf_fwd_wmma(
    const float* __restrict__ unary,    // [T, B, N]
    const float* __restrict__ trans,    // [1, N, N]  trans[to*N + fr]
    const int*   __restrict__ lengths,  // [B]
    float*       __restrict__ out)      // [B]
{
    __shared__ __align__(16) _Float16 s_p[16][CRF_N]; // w (row max == 1), f16
    __shared__ __align__(16) float    s_v[16][CRF_N]; // candidate v, f32
    __shared__ float    s_part[16][8];                // partial row maxes
    __shared__ float    s_C[16];                      // per-row log2 shift

    const int tid  = threadIdx.x;   // 0..127
    const int wave = tid >> 5;      // "to" tile
    const int lane = tid & 31;
    const int lg   = lane >> 4;
    const int ln   = lane & 15;
    const int b_base = blockIdx.x * 16;

    // ---- constant B fragments: E[fr][to] = exp(trans[to][fr]) (f16) ----
    // 16-bit B (32x16): N = lane&15; half j: K = 16*lanegroup + j (order chosen here, same as A).
    v16h bf0, bf1;
    {
        const int to = wave * 16 + ln;
        #pragma unroll
        for (int j = 0; j < 16; ++j) {
            const int fr = lg * 16 + j;
            bf0[j] = (_Float16)__builtin_amdgcn_exp2f(trans[to*CRF_N + fr     ] * LOG2E);
            bf1[j] = (_Float16)__builtin_amdgcn_exp2f(trans[to*CRF_N + fr + 32] * LOG2E);
        }
    }

    // C layout: reg v -> row v + 8*lanegroup, col = wave*16 + (lane&15)
    int u_off[8];
    #pragma unroll
    for (int v = 0; v < 8; ++v)
        u_off[v] = (b_base + v + 8*lg) * CRF_N + wave*16 + ln;

    const int rrow   = tid >> 3;     // 0..15, 8 threads per row
    const int rslot  = tid & 7;
    const int lenrow = lengths[b_base + rrow];

    // ---- init: w = delta(START_IDX), C = 0 ----
    for (int e = tid; e < 16 * CRF_N; e += 128)
        s_p[e >> 6][e & 63] = (_Float16)(((e & 63) == START_IDX) ? 1.0f : 0.0f);
    if (tid < 16) s_C[tid] = 0.0f;
    __syncthreads();

    for (int t = 0; t < CRF_T; ++t) {
        // unary loads issued early to overlap with the WMMAs
        const float* u_t = unary + (size_t)t * (CRF_B * CRF_N);
        float u[8];
        #pragma unroll
        for (int v = 0; v < 8; ++v) u[v] = u_t[u_off[v]];

        if (t + 1 < CRF_T)
            __builtin_prefetch(u_t + CRF_B*CRF_N + (size_t)b_base*CRF_N + tid*8, 0, 0);

        // ---- A fragments (K = 16*lanegroup + j, matching B; lowers to b128) ----
        v16h a0, a1;
        #pragma unroll
        for (int j = 0; j < 16; ++j) {
            const int kk = (lg << 4) + j;
            a0[j] = s_p[ln][kk];
            a1[j] = s_p[ln][kk + 32];
        }

        // ---- q = w x E : two chained WMMAs over K = 64 ----
        v8f c = {};
        c = __builtin_amdgcn_wmma_f32_16x16x32_f16(
                false, a0, false, bf0, (short)0, c, false, false);
        c = __builtin_amdgcn_wmma_f32_16x16x32_f16(
                false, a1, false, bf1, (short)0, c, false, false);

        // ---- v = q * 2^(unary * log2e)  (linear domain, no logs) ----
        {
            const int to = wave*16 + ln;
            #pragma unroll
            for (int v = 0; v < 8; ++v)
                s_v[v + 8*lg][to] = c[v] * __builtin_amdgcn_exp2f(u[v] * LOG2E);
        }
        __syncthreads();

        // ---- partial row max (values kept in regs for the commit) ----
        float ch[8];
        #pragma unroll
        for (int k = 0; k < 8; ++k) ch[k] = s_v[rrow][rslot*8 + k];
        float m = ch[0];
        #pragma unroll
        for (int k = 1; k < 8; ++k) m = fmaxf(m, ch[k]);
        s_part[rrow][rslot] = m;
        __syncthreads();

        // ---- commit (uniform per row): normalize w, bump C by log2(max) ----
        if (t < lenrow) {
            float mv = s_part[rrow][0];
            #pragma unroll
            for (int k = 1; k < 8; ++k) mv = fmaxf(mv, s_part[rrow][k]);
            const float r = __builtin_amdgcn_rcpf(mv);
            #pragma unroll
            for (int k = 0; k < 8; ++k)
                s_p[rrow][rslot*8 + k] = (_Float16)(ch[k] * r);
            if (rslot == 0)
                s_C[rrow] += __builtin_amdgcn_logf(mv);   // v_log_f32 = log2
        }
        __syncthreads();
    }

    // ---- terminal: out[b] = ln2 * (C_b + log2( sum_fr w * 2^(tr_end*log2e) ))
    if (tid < 16) {
        const int b = tid;
        float S = 0.0f;
        #pragma unroll 4
        for (int fr = 0; fr < CRF_N; ++fr)
            S += (float)s_p[b][fr] *
                 __builtin_amdgcn_exp2f(trans[END_IDX*CRF_N + fr] * LOG2E);
        out[b_base + b] = (s_C[b] + __builtin_amdgcn_logf(S)) * LN2;
    }
}

extern "C" void kernel_launch(void* const* d_in, const int* in_sizes, int n_in,
                              void* d_out, int out_size, void* d_ws, size_t ws_size,
                              hipStream_t stream) {
    const float* unary   = (const float*)d_in[0];   // [T, B, N] f32
    const float* trans   = (const float*)d_in[1];   // [1, N, N] f32
    const int*   lengths = (const int*)d_in[2];     // [B] i32
    float*       out     = (float*)d_out;           // [B] f32

    crf_fwd_wmma<<<dim3(CRF_B / 16), dim3(128), 0, stream>>>(
        unary, trans, lengths, out);
}
